// Transolver_17068200034966
// MI455X (gfx1250) — matrix operand order repaired and checked by reading, not compiled, not run on hardware
//
#include <hip/hip_runtime.h>
#include <hip/hip_bf16.h>
#include <math.h>

// ---------------- CDNA5 WMMA plumbing ----------------
typedef __attribute__((ext_vector_type(16))) __bf16 v16bf;
typedef __attribute__((ext_vector_type(8)))  float  v8f;

union FragAB {
  v16bf v;
  __bf16 u[16];
  uint4 q[2];
};

__device__ __forceinline__ v8f wmma_bf16(const FragAB& a, const FragAB& b, v8f c) {
  return __builtin_amdgcn_wmma_f32_16x16x32_bf16(false, a.v, false, b.v, (short)0, c, false, false);
}

// A fragment [16 x 32] from row-major bf16 buffer T (stride S), tile (row0,k0).
// Lane l: row = row0 + (l&15); half h = l>>4; elems 0..15 -> K = k0 + h*16 + e
// (contiguous order, matching packed B)
__device__ __forceinline__ FragAB load_a(const __bf16* T, int S, int row0, int k0, int lane) {
  FragAB f;
  const __bf16* p = T + (row0 + (lane & 15)) * S + k0;
  int h16 = (lane >> 4) * 16;
#pragma unroll
  for (int e = 0; e < 16; ++e) { f.u[e] = p[h16 + e]; }
  return f;
}
// B fragment from packed weights: slab = (n_tile * K/32 + k_tile); per-lane 32B contiguous.
__device__ __forceinline__ FragAB load_b_pk(const __bf16* P, int slab, int lane) {
  FragAB f;
  const uint4* p = (const uint4*)(P + ((long)slab * 512 + lane * 16));
  f.q[0] = p[0];
  f.q[1] = p[1];
  return f;
}
// B fragment [32 x 16] with B[k][n] = T[(k0+k)*S + n0+n] (activation, K along rows)
__device__ __forceinline__ FragAB load_b_act(const __bf16* T, int S, int k0, int n0, int lane) {
  FragAB f;
  int n = n0 + (lane & 15);
  int kb = k0 + ((lane >> 4) << 4);
#pragma unroll
  for (int e = 0; e < 16; ++e) f.u[e] = T[(kb + e) * S + n];
  return f;
}

__device__ __forceinline__ float wsum(float v) {
#pragma unroll
  for (int m = 16; m; m >>= 1) v += __shfl_xor(v, m, 32);
  return v;
}
__device__ __forceinline__ float geluf(float x) {
  return 0.5f * x * (1.0f + erff(x * 0.70710678118654752f));
}

// ---------------- problem constants ----------------
#define BZ 4
#define NN 32768
#define DD 256
#define HH 8
#define GG 32
#define DHD 32
#define MLPH 1024
#define MR ((long)BZ * NN)   // 131072 rows

// ---------------- tiny kernels ----------------
// cast + pack weights [Nout x K] fp32 into WMMA-B fragment order:
// slab s = n_tile*(K/32)+k_tile; within slab, lane l holds 16 contiguous K elems
// of row n_tile*16+(l&15), K base k_tile*32 + (l>=16?16:0).
__global__ void __launch_bounds__(256) k_pack(const float* __restrict__ W,
                                              __bf16* __restrict__ P, int Nout, int K) {
  int gid = blockIdx.x * 256 + threadIdx.x;   // one 16-element group
  int total = (Nout * K) >> 4;
  if (gid >= total) return;
  int s = gid >> 5, l = gid & 31;
  int kt = K >> 5;
  int nt = s / kt, ks = s % kt;
  int n = nt * 16 + (l & 15);
  int kb = ks * 32 + ((l >> 4) << 4);
  const float* src = W + (long)n * K + kb;
  __bf16* d = P + (long)gid * 16;
#pragma unroll
  for (int e = 0; e < 16; ++e) d[e] = (__bf16)src[e];
}
__global__ void __launch_bounds__(256) k_zero(float* __restrict__ p, long n) {
  long i = (long)blockIdx.x * 256 + threadIdx.x;
  if (i < n) p[i] = 0.0f;
}

// spatial-bias MLP: 4 -> 64 -> 64 -> 32 (exact GELU), one row per thread
__global__ void __launch_bounds__(256) k_sb(const float* __restrict__ xy,
    const float* __restrict__ W1, const float* __restrict__ b1,
    const float* __restrict__ W2, const float* __restrict__ b2,
    const float* __restrict__ W3, const float* __restrict__ b3,
    float* __restrict__ sb) {
  long row = (long)blockIdx.x * 256 + threadIdx.x;
  if (row >= MR) return;
  float x0 = xy[row*4+0], x1 = xy[row*4+1], x2 = xy[row*4+2], x3 = xy[row*4+3];
  float h1[64];
#pragma unroll 8
  for (int i = 0; i < 64; ++i) {
    const float* w = W1 + i*4;
    h1[i] = geluf(w[0]*x0 + w[1]*x1 + w[2]*x2 + w[3]*x3 + b1[i]);
  }
  float h2[64];
  for (int i = 0; i < 64; ++i) {
    float s = b2[i]; const float* w = W2 + i*64;
#pragma unroll 8
    for (int j = 0; j < 64; ++j) s += w[j]*h1[j];
    h2[i] = geluf(s);
  }
  for (int g = 0; g < 32; ++g) {
    float s = b3[g]; const float* w = W3 + g*64;
#pragma unroll 8
    for (int j = 0; j < 64; ++j) s += w[j]*h2[j];
    sb[row*32+g] = s;
  }
}

// LN of fx -> bf16 (one wave per row, 8 elems per lane)
__global__ void __launch_bounds__(256) k_ln_in(const float* __restrict__ x,
    const float* __restrict__ g, const float* __restrict__ b,
    __bf16* __restrict__ out) {
  long row = (long)blockIdx.x * 8 + (threadIdx.x >> 5);
  int lane = threadIdx.x & 31;
  const float* xr = x + row * DD;
  float v[8]; float s = 0.f;
#pragma unroll
  for (int j = 0; j < 8; ++j) { v[j] = xr[lane + 32*j]; s += v[j]; }
  s = wsum(s); float mean = s * (1.f/256.f);
  float q = 0.f;
#pragma unroll
  for (int j = 0; j < 8; ++j) { float d = v[j]-mean; q += d*d; }
  q = wsum(q); float rs = rsqrtf(q * (1.f/256.f) + 1e-5f);
#pragma unroll
  for (int j = 0; j < 8; ++j) {
    int d = lane + 32*j;
    out[row*DD + d] = (__bf16)((v[j]-mean)*rs*g[d] + b[d]);
  }
}

// ---------------- fused slice kernel ----------------
// Per 256-row block (8 iters of 32 rows):
//   x_mid = h@Wx^T+bx (WMMA) -> logits per head (WMMA, K=DH=32)
//   softmax over G=32 in registers (shfl butterflies), sw -> global + LDS
//   fx_mid = h@Wfx^T+bfx (WMMA); tok += sw^T @ fx_mid (WMMA), atomics at end
__global__ void __launch_bounds__(256) k_slice(
    const __bf16* __restrict__ hbf,
    const __bf16* __restrict__ Wx_p,  const float* __restrict__ b_x,
    const __bf16* __restrict__ Wfx_p, const float* __restrict__ b_fx,
    const __bf16* __restrict__ Wsl_p, const float* __restrict__ b_slice,
    const float* __restrict__ temperature, const float* __restrict__ sb,
    __bf16* __restrict__ sw_out, float* __restrict__ tok,
    float* __restrict__ snorm) {
  __shared__ __bf16 pool[3][32*256];   // 48 KB
  __bf16* sh_h  = pool[0];             // also reused as per-wave sw scratch
  __bf16* sh_xm = pool[1];
  __bf16* sh_fm = pool[2];

  const int tid = threadIdx.x, w = tid >> 5, lane = tid & 31;
  const int blk = blockIdx.x;
  const int bb = (int)(((long)blk * 256) / NN);
  const float invT = 1.0f / temperature[w];
  const float bsl0 = b_slice[lane & 15];
  const float bsl1 = b_slice[16 + (lane & 15)];
  const int hl = lane >> 4;

  // phase-1 wave assignment: fixed (gemm, row-tile), 8 consecutive col tiles
  const int which = w >> 2;            // 0: x_mid (Wx), 1: fx_mid (Wfx)
  const int rt1   = (w >> 1) & 1;
  const int ct0   = (w & 1) * 8;
  const __bf16* Wp = which ? Wfx_p : Wx_p;
  const float* bias = which ? b_fx : b_x;
  __bf16* dst = which ? sh_fm : sh_xm;

  // invariant logits B fragments (W_slice packed: slabs 0,1)
  FragAB bsl_f0 = load_b_pk(Wsl_p, 0, lane);
  FragAB bsl_f1 = load_b_pk(Wsl_p, 1, lane);

  v8f tokacc[2][2];
#pragma unroll
  for (int a = 0; a < 2; ++a)
#pragma unroll
    for (int c = 0; c < 2; ++c) { v8f z = {}; tokacc[a][c] = z; }
  float sn0 = 0.f, sn1 = 0.f;

  for (int it = 0; it < 8; ++it) {
    __syncthreads();                                 // protect pool reuse
    const long r0 = (long)blk * 256 + it * 32;
    {                                                // coop load 32x256 bf16 tile
      const uint4* src = (const uint4*)(hbf + r0 * DD);
      uint4* dl = (uint4*)sh_h;
#pragma unroll
      for (int i = 0; i < 4; ++i) dl[tid + i*256] = src[tid + i*256];
    }
    __syncthreads();
    // ---- phase 1: one A per k-step shared by 8 col tiles, 8 C accumulators ----
    {
      v8f c[8];
#pragma unroll
      for (int i = 0; i < 8; ++i) {
        float bv = bias[(ct0 + i)*16 + (lane & 15)];
        v8f z = {};
#pragma unroll
        for (int v = 0; v < 8; ++v) z[v] = bv;
        c[i] = z;
      }
#pragma unroll 1
      for (int ks = 0; ks < 8; ++ks) {
        FragAB a = load_a(sh_h, DD, rt1*16, ks*32, lane);
#pragma unroll
        for (int i = 0; i < 8; ++i) {
          FragAB b = load_b_pk(Wp, (ct0 + i)*8 + ks, lane);
          c[i] = wmma_bf16(a, b, c[i]);
        }
      }
#pragma unroll
      for (int i = 0; i < 8; ++i) {
        int col = (ct0 + i)*16 + (lane & 15);
#pragma unroll
        for (int v = 0; v < 8; ++v)
          dst[(rt1*16 + v + 8*hl)*DD + col] = (__bf16)c[i][v];
      }
    }
    __syncthreads();
    // ---- phase 2: wave w <-> head w; sh_h region reused as per-wave sw scratch ----
    __bf16* swsh = pool[0] + w * 1024;   // 32 x 32
    const int g0 = lane & 15, g1 = 16 + g0;
#pragma unroll
    for (int rt = 0; rt < 2; ++rt) {
      v8f L0 = {}; v8f L1 = {};
      FragAB a = load_a(sh_xm, DD, rt*16, w*32, lane);           // K = DH = 32
      L0 = wmma_bf16(a, bsl_f0, L0);
      L1 = wmma_bf16(a, bsl_f1, L1);
#pragma unroll
      for (int v = 0; v < 8; ++v) {
        long rowg = r0 + rt*16 + v + 8*hl;
        float l0 = (L0[v] + bsl0) * invT + 0.1f * sb[rowg*GG + g0];
        float l1 = (L1[v] + bsl1) * invT + 0.1f * sb[rowg*GG + g1];
        float m = fmaxf(l0, l1);
#pragma unroll
        for (int off = 8; off; off >>= 1) m = fmaxf(m, __shfl_xor(m, off, 32));
        float e0 = __expf(l0 - m), e1 = __expf(l1 - m);
        float s = e0 + e1;
#pragma unroll
        for (int off = 8; off; off >>= 1) s += __shfl_xor(s, off, 32);
        float inv = 1.0f / s;
        float s0 = e0 * inv, s1 = e1 * inv;
        sn0 += s0; sn1 += s1;
        __bf16 q0 = (__bf16)s0, q1 = (__bf16)s1;
        sw_out[rowg*(HH*GG) + w*GG + g0] = q0;
        sw_out[rowg*(HH*GG) + w*GG + g1] = q1;
        int nloc = rt*16 + v + 8*hl;
        swsh[nloc*32 + g0] = q0;
        swsh[nloc*32 + g1] = q1;
      }
    }
    // tok partial: A[g][n] = sw[n][g], B[n][c] = fx_mid[n][w*32+c], K = 32 rows
#pragma unroll
    for (int gt = 0; gt < 2; ++gt) {
      FragAB a;
      int g = gt*16 + (lane & 15);
      int n16 = (lane >> 4) * 16;
#pragma unroll
      for (int e = 0; e < 16; ++e) {
        a.u[e]     = swsh[(n16 + e)*32 + g];
      }
#pragma unroll
      for (int c2 = 0; c2 < 2; ++c2) {
        FragAB b = load_b_act(sh_fm, DD, 0, w*32 + c2*16, lane);
        tokacc[gt][c2] = wmma_bf16(a, b, tokacc[gt][c2]);
      }
    }
  }
  // flush partials
#pragma unroll
  for (int gt = 0; gt < 2; ++gt)
#pragma unroll
    for (int c2 = 0; c2 < 2; ++c2) {
      int c = c2*16 + (lane & 15);
#pragma unroll
      for (int v = 0; v < 8; ++v) {
        int g = gt*16 + v + 8*hl;
        atomicAdd(&tok[(((long)bb*HH + w)*GG + g)*DHD + c], tokacc[gt][c2][v]);
      }
    }
  float t0 = sn0 + __shfl_xor(sn0, 16, 32);
  float t1 = sn1 + __shfl_xor(sn1, 16, 32);
  if (lane < 16) {
    atomicAdd(&snorm[((long)bb*HH + w)*GG + lane],      t0);
    atomicAdd(&snorm[((long)bb*HH + w)*GG + 16 + lane], t1);
  }
}

// ---------------- token attention (tiny, one block per batch) ----------------
__global__ void __launch_bounds__(256) k_attn(
    const float* __restrict__ tok, const float* __restrict__ snorm,
    const float* __restrict__ Wq, const float* __restrict__ Wk,
    const float* __restrict__ Wv, const float* __restrict__ scale,
    const float* __restrict__ slice_res, float* __restrict__ ot) {
  __shared__ float tf[HH*GG*DHD];                       // 32 KB
  __shared__ float kv[GG*DHD], kk[GG*DHD], vv[GG*DHD], qn[GG*DHD], kn[GG*DHD], aa[GG*GG];
  const int b = blockIdx.x, tid = threadIdx.x;
  for (int i = tid; i < HH*GG*DHD; i += 256) {
    int h = i >> 10, g = (i >> 5) & 31;
    tf[i] = tok[(long)b*HH*GG*DHD + i] / (snorm[(b*HH + h)*GG + g] + 1e-5f);
  }
  __syncthreads();
  for (int i = tid; i < GG*DHD; i += 256) {
    float s = 0.f;
    for (int h = 0; h < HH; ++h) s += tf[h*1024 + i];
    kv[i] = s * (1.0f / HH);
  }
  __syncthreads();
  for (int i = tid; i < GG*DHD; i += 256) {
    int g = i >> 5, c = i & 31;
    float sk = 0.f, sv = 0.f;
    for (int d = 0; d < DHD; ++d) { float t = kv[g*32+d]; sk += t*Wk[c*32+d]; sv += t*Wv[c*32+d]; }
    kk[i] = sk; vv[i] = sv;
  }
  __syncthreads();
  if (tid < GG) {
    float s = 0.f;
    for (int c = 0; c < DHD; ++c) { float t = kk[tid*32+c]; s += t*t; }
    float inv = 1.0f / fmaxf(sqrtf(s), 1e-12f);
    for (int c = 0; c < DHD; ++c) kn[tid*32+c] = kk[tid*32+c] * inv;
  }
  __syncthreads();
  const float sres = slice_res[0];
  for (int h = 0; h < HH; ++h) {
    for (int i = tid; i < GG*DHD; i += 256) {
      int g = i >> 5, c = i & 31;
      float s = 0.f;
      for (int d = 0; d < DHD; ++d) s += tf[h*1024 + g*32 + d] * Wq[c*32+d];
      qn[i] = s;
    }
    __syncthreads();
    if (tid < GG) {
      float s = 0.f;
      for (int c = 0; c < DHD; ++c) { float t = qn[tid*32+c]; s += t*t; }
      float inv = 1.0f / fmaxf(sqrtf(s), 1e-12f);
      for (int c = 0; c < DHD; ++c) qn[tid*32+c] *= inv;
    }
    __syncthreads();
    for (int i = tid; i < GG*GG; i += 256) {
      int g = i >> 5, j = i & 31;
      float s = 0.f;
      for (int d = 0; d < DHD; ++d) s += qn[g*32+d] * kn[j*32+d];
      aa[i] = s * scale[h];
    }
    __syncthreads();
    if (tid < GG) {
      float m = -1e30f;
      for (int j = 0; j < GG; ++j) m = fmaxf(m, aa[tid*32+j]);
      float s = 0.f;
      for (int j = 0; j < GG; ++j) { float e = __expf(aa[tid*32+j]-m); aa[tid*32+j] = e; s += e; }
      float inv = 1.0f / s;
      for (int j = 0; j < GG; ++j) aa[tid*32+j] *= inv;
    }
    __syncthreads();
    for (int i = tid; i < GG*DHD; i += 256) {
      int g = i >> 5, c = i & 31;
      float s = 0.f;
      for (int j = 0; j < GG; ++j) s += aa[g*32+j] * vv[j*32+c];
      ot[(long)b*HH*GG*DHD + h*1024 + i] = s + sres * tf[h*1024 + i];
    }
    __syncthreads();
  }
}

// scatter back: y[n][h*32+c] = sum_g ot[b,h,g,c]*sw[n,h,g]; one (row,head) per thread
__global__ void __launch_bounds__(256) k_scatter(
    const __bf16* __restrict__ sw, const float* __restrict__ ot,
    __bf16* __restrict__ y) {
  long t = (long)blockIdx.x * 256 + threadIdx.x;
  long row = t >> 3; int h = (int)(t & 7);
  int b = (int)(row >> 15);
  float acc[32];
#pragma unroll
  for (int c = 0; c < 32; ++c) acc[c] = 0.f;
  const __bf16* swr = sw + row*(HH*GG) + h*GG;
  const float* otb = ot + ((long)b*HH + h)*GG*DHD;
#pragma unroll 4
  for (int g = 0; g < GG; ++g) {
    float s = (float)swr[g];
    const float* o = otb + g*DHD;
#pragma unroll
    for (int c = 0; c < 32; ++c) acc[c] += s * o[c];
  }
  __bf16* yr = y + row*DD + h*DHD;
#pragma unroll
  for (int c = 0; c < 32; ++c) yr[c] = (__bf16)acc[c];
}

// out-projection + residual + LN1p (-> fx2 fp32) + LN2 (-> bf16 MLP input)
__global__ void __launch_bounds__(256) k_outproj(
    const __bf16* __restrict__ y, const __bf16* __restrict__ Wo_p,
    const float* __restrict__ bo, const float* __restrict__ fx_in,
    const float* __restrict__ g1p, const float* __restrict__ b1p,
    const float* __restrict__ g2, const float* __restrict__ b2,
    float* __restrict__ fx2, __bf16* __restrict__ m_in) {
  __shared__ __bf16 sh_in[32*256];   // 16 KB
  __shared__ float sh_t[32*256];     // 32 KB
  const int tid = threadIdx.x, w = tid >> 5, lane = tid & 31, hl = lane >> 4;
  const long r0 = (long)blockIdx.x * 32;
  const int rt = w >> 2;             // tiles t = w*4+i -> same row tile per wave
  const int ct0 = (w & 3) * 4;
  {
    const uint4* src = (const uint4*)(y + r0 * DD);
    uint4* dl = (uint4*)sh_in;
#pragma unroll
    for (int i = 0; i < 4; ++i) dl[tid + i*256] = src[tid + i*256];
  }
  __syncthreads();
  {
    v8f c[4];
#pragma unroll
    for (int i = 0; i < 4; ++i) {
      float bv = bo[(ct0 + i)*16 + (lane & 15)];
      v8f z = {};
#pragma unroll
      for (int v = 0; v < 8; ++v) z[v] = bv;
      c[i] = z;
    }
#pragma unroll 1
    for (int ks = 0; ks < 8; ++ks) {
      FragAB a = load_a(sh_in, DD, rt*16, ks*32, lane);
#pragma unroll
      for (int i = 0; i < 4; ++i) {
        FragAB b = load_b_pk(Wo_p, (ct0 + i)*8 + ks, lane);
        c[i] = wmma_bf16(a, b, c[i]);
      }
    }
#pragma unroll
    for (int i = 0; i < 4; ++i) {
      int col = (ct0 + i)*16 + (lane & 15);
#pragma unroll
      for (int v = 0; v < 8; ++v) {
        int rl = rt*16 + v + 8*hl;
        sh_t[rl*DD + col] = c[i][v] + fx_in[(r0 + rl)*DD + col];
      }
    }
  }
  __syncthreads();
  for (int rr = 0; rr < 4; ++rr) {
    int rl = w*4 + rr;
    float v[8]; float s = 0.f;
#pragma unroll
    for (int j = 0; j < 8; ++j) { v[j] = sh_t[rl*DD + lane + 32*j]; s += v[j]; }
    s = wsum(s); float mean = s * (1.f/256.f);
    float q = 0.f;
#pragma unroll
    for (int j = 0; j < 8; ++j) { float d = v[j]-mean; q += d*d; }
    q = wsum(q); float rs = rsqrtf(q * (1.f/256.f) + 1e-5f);
    float f2[8]; float s2 = 0.f;
#pragma unroll
    for (int j = 0; j < 8; ++j) {
      int d = lane + 32*j;
      f2[j] = (v[j]-mean)*rs*g1p[d] + b1p[d];
      s2 += f2[j];
      fx2[(r0 + rl)*DD + d] = f2[j];
    }
    s2 = wsum(s2); float m2 = s2 * (1.f/256.f);
    float q2 = 0.f;
#pragma unroll
    for (int j = 0; j < 8; ++j) { float d = f2[j]-m2; q2 += d*d; }
    q2 = wsum(q2); float rs2 = rsqrtf(q2 * (1.f/256.f) + 1e-5f);
#pragma unroll
    for (int j = 0; j < 8; ++j) {
      int d = lane + 32*j;
      m_in[(r0 + rl)*DD + d] = (__bf16)((f2[j]-m2)*rs2*g2[d] + b2[d]);
    }
  }
}

// fused MLP (D->4D GELU ->D) + residual + LN2p -> d_out; SE sums via atomics
__global__ void __launch_bounds__(256) k_mlp(
    const __bf16* __restrict__ min_, const __bf16* __restrict__ W1_p,
    const float* __restrict__ b1, const __bf16* __restrict__ W2_p,
    const float* __restrict__ b2, const float* __restrict__ fx2,
    const float* __restrict__ g2p, const float* __restrict__ b2p,
    float* __restrict__ outp, float* __restrict__ se_sum) {
  __shared__ __bf16 pool[2][32*256];    // 32 KB (m_in tile | u tile, later fp32 t)
  __shared__ float sh_se[256];
  __bf16* sh_in = pool[0];
  __bf16* sh_u  = pool[1];
  float* sh_t = (float*)pool;
  const int tid = threadIdx.x, w = tid >> 5, lane = tid & 31, hl = lane >> 4;
  const long r0 = (long)blockIdx.x * 32;
  const int b = (int)(r0 >> 15);
  const int rt = w >> 2;
  const int ct0 = (w & 3) * 4;
  sh_se[tid] = 0.0f;
  {
    const uint4* src = (const uint4*)(min_ + r0 * DD);
    uint4* dl = (uint4*)sh_in;
#pragma unroll
    for (int i = 0; i < 4; ++i) dl[tid + i*256] = src[tid + i*256];
  }
  __syncthreads();
  v8f yacc[4];
#pragma unroll
  for (int i = 0; i < 4; ++i) {
    float bv = b2[(ct0 + i)*16 + (lane & 15)];
    v8f z = {};
#pragma unroll
    for (int v = 0; v < 8; ++v) z[v] = bv;
    yacc[i] = z;
  }
#pragma unroll 1
  for (int ch = 0; ch < 4; ++ch) {
    __builtin_prefetch(W1_p + ((long)(ch*16 + ct0) * 8) * 512, 0, 1);
    {
      v8f c[4];
#pragma unroll
      for (int i = 0; i < 4; ++i) {
        float bv = b1[ch*256 + (ct0 + i)*16 + (lane & 15)];
        v8f z = {};
#pragma unroll
        for (int v = 0; v < 8; ++v) z[v] = bv;
        c[i] = z;
      }
#pragma unroll 1
      for (int ks = 0; ks < 8; ++ks) {
        FragAB a = load_a(sh_in, DD, rt*16, ks*32, lane);
#pragma unroll
        for (int i = 0; i < 4; ++i) {
          FragAB b = load_b_pk(W1_p, (ch*16 + ct0 + i)*8 + ks, lane);
          c[i] = wmma_bf16(a, b, c[i]);
        }
      }
#pragma unroll
      for (int i = 0; i < 4; ++i) {
        int col = (ct0 + i)*16 + (lane & 15);
#pragma unroll
        for (int v = 0; v < 8; ++v)
          sh_u[(rt*16 + v + 8*hl)*DD + col] = (__bf16)geluf(c[i][v]);
      }
    }
    __syncthreads();
#pragma unroll 1
    for (int ks = 0; ks < 8; ++ks) {
      FragAB a = load_a(sh_u, DD, rt*16, ks*32, lane);
#pragma unroll
      for (int i = 0; i < 4; ++i) {
        FragAB b = load_b_pk(W2_p, (ct0 + i)*32 + ch*8 + ks, lane);
        yacc[i] = wmma_bf16(a, b, yacc[i]);
      }
    }
    __syncthreads();
  }
#pragma unroll
  for (int i = 0; i < 4; ++i) {
    int col = (ct0 + i)*16 + (lane & 15);
#pragma unroll
    for (int v = 0; v < 8; ++v) {
      int rl = rt*16 + v + 8*hl;
      sh_t[rl*DD + col] = yacc[i][v] + fx2[(r0 + rl)*DD + col];
    }
  }
  __syncthreads();
  for (int rr = 0; rr < 4; ++rr) {
    int rl = w*4 + rr;
    float v[8]; float s = 0.f;
#pragma unroll
    for (int j = 0; j < 8; ++j) { v[j] = sh_t[rl*DD + lane + 32*j]; s += v[j]; }
    s = wsum(s); float mean = s * (1.f/256.f);
    float q = 0.f;
#pragma unroll
    for (int j = 0; j < 8; ++j) { float d = v[j]-mean; q += d*d; }
    q = wsum(q); float rs = rsqrtf(q * (1.f/256.f) + 1e-5f);
#pragma unroll
    for (int j = 0; j < 8; ++j) {
      int d = lane + 32*j;
      float o = (v[j]-mean)*rs*g2p[d] + b2p[d];
      outp[(r0 + rl)*DD + d] = o;
      atomicAdd(&sh_se[d], o);
    }
  }
  __syncthreads();
  atomicAdd(&se_sum[b*DD + tid], sh_se[tid]);
}

// SE: mean -> 64 (gelu) -> 256 (sigmoid); one block
__global__ void __launch_bounds__(256) k_se(const float* __restrict__ se_sum,
    const float* __restrict__ W1, const float* __restrict__ b1,
    const float* __restrict__ W2, const float* __restrict__ b2,
    float* __restrict__ se) {
  __shared__ float mean[BZ*DD];
  __shared__ float hid[BZ*64];
  int tid = threadIdx.x;
  for (int i = tid; i < BZ*DD; i += 256) mean[i] = se_sum[i] * (1.0f/32768.0f);
  __syncthreads();
  {
    int b = tid >> 6, s = tid & 63;
    float acc = b1[s];
    for (int d = 0; d < DD; ++d) acc += mean[b*DD + d] * W1[s*DD + d];
    hid[tid] = geluf(acc);
  }
  __syncthreads();
  for (int i = tid; i < BZ*DD; i += 256) {
    int b = i >> 8, d = i & 255;
    float acc = b2[d];
    for (int s = 0; s < 64; ++s) acc += hid[b*64 + s] * W2[d*64 + s];
    se[i] = 1.0f / (1.0f + __expf(-acc));
  }
}

__global__ void __launch_bounds__(256) k_scale(float* __restrict__ out,
                                               const float* __restrict__ se) {
  const long total = MR * DD;
  for (long i = (long)blockIdx.x * 256 + threadIdx.x; i < total; i += (long)gridDim.x * 256) {
    long row = i >> 8; int d = (int)(i & 255); int b = (int)(row >> 15);
    out[i] *= se[b*DD + d];
  }
}

// ---------------- host launch ----------------
extern "C" void kernel_launch(void* const* d_in, const int* in_sizes, int n_in,
                              void* d_out, int out_size, void* d_ws, size_t ws_size,
                              hipStream_t stream) {
  (void)in_sizes; (void)n_in; (void)out_size; (void)ws_size;
  // input order per setup_inputs()
  const float* fx       = (const float*)d_in[0];
  const float* raw_xy   = (const float*)d_in[1];
  const float* W_fx     = (const float*)d_in[2];
  const float* b_fx     = (const float*)d_in[3];
  const float* W_x      = (const float*)d_in[4];
  const float* b_x      = (const float*)d_in[5];
  const float* W_slice  = (const float*)d_in[6];
  const float* b_slice  = (const float*)d_in[7];
  const float* temperat = (const float*)d_in[8];
  const float* Wq       = (const float*)d_in[9];
  const float* Wk       = (const float*)d_in[10];
  const float* Wv       = (const float*)d_in[11];
  const float* attn_sc  = (const float*)d_in[12];
  const float* slice_res= (const float*)d_in[13];
  const float* W_out    = (const float*)d_in[14];
  const float* b_out    = (const float*)d_in[15];
  const float* ln1_g    = (const float*)d_in[16];
  const float* ln1_b    = (const float*)d_in[17];
  const float* ln1p_g   = (const float*)d_in[18];
  const float* ln1p_b   = (const float*)d_in[19];
  const float* ln2_g    = (const float*)d_in[20];
  const float* ln2_b    = (const float*)d_in[21];
  const float* ln2p_g   = (const float*)d_in[22];
  const float* ln2p_b   = (const float*)d_in[23];
  const float* Wm1      = (const float*)d_in[24];
  const float* bm1      = (const float*)d_in[25];
  const float* Wm2      = (const float*)d_in[26];
  const float* bm2      = (const float*)d_in[27];
  const float* Wsb1     = (const float*)d_in[28];
  const float* bsb1     = (const float*)d_in[29];
  const float* Wsb2     = (const float*)d_in[30];
  const float* bsb2     = (const float*)d_in[31];
  const float* Wsb3     = (const float*)d_in[32];
  const float* bsb3     = (const float*)d_in[33];
  const float* Wse1     = (const float*)d_in[34];
  const float* bse1     = (const float*)d_in[35];
  const float* Wse2     = (const float*)d_in[36];
  const float* bse2     = (const float*)d_in[37];
  float* out = (float*)d_out;

  // workspace carve-up (256B aligned)
  char* base = (char*)d_ws;
  size_t off = 0;
  auto take = [&](size_t bytes) -> char* {
    char* p = base + off;
    off += (bytes + 255) & ~(size_t)255;
    return p;
  };
  __bf16* wfx_p  = (__bf16*)take(65536*2);
  __bf16* wx_p   = (__bf16*)take(65536*2);
  __bf16* wo_p   = (__bf16*)take(65536*2);
  __bf16* wm1_p  = (__bf16*)take(262144*2);
  __bf16* wm2_p  = (__bf16*)take(262144*2);
  __bf16* wsl_p  = (__bf16*)take(1024*2);
  float*  sb     = (float*)take((size_t)MR*GG*4);
  __bf16* hbf    = (__bf16*)take((size_t)MR*DD*2);
  __bf16* swbuf  = (__bf16*)take((size_t)MR*HH*GG*2);
  float*  tok    = (float*)take((size_t)BZ*HH*GG*DHD*4);
  float*  snorm  = (float*)take((size_t)BZ*HH*GG*4);
  float*  ot     = (float*)take((size_t)BZ*HH*GG*DHD*4);
  __bf16* ybuf   = (__bf16*)take((size_t)MR*DD*2);
  float*  fx2    = (float*)take((size_t)MR*DD*4);
  __bf16* minbuf = (__bf16*)take((size_t)MR*DD*2);
  float*  sesum  = (float*)take((size_t)BZ*DD*4);
  float*  sebuf  = (float*)take((size_t)BZ*DD*4);

  // 0) weight cast+pack into WMMA fragment order, accumulator zeroing
  k_pack<<<dim3(16),  dim3(256), 0, stream>>>(W_x,     wx_p,  256, 256);
  k_pack<<<dim3(16),  dim3(256), 0, stream>>>(W_fx,    wfx_p, 256, 256);
  k_pack<<<dim3(16),  dim3(256), 0, stream>>>(W_out,   wo_p,  256, 256);
  k_pack<<<dim3(64),  dim3(256), 0, stream>>>(Wm1,     wm1_p, 1024, 256);
  k_pack<<<dim3(64),  dim3(256), 0, stream>>>(Wm2,     wm2_p, 256, 1024);
  k_pack<<<dim3(1),   dim3(256), 0, stream>>>(W_slice, wsl_p, 32, 32);
  k_zero<<<dim3(128), dim3(256), 0, stream>>>(tok,   (long)BZ*HH*GG*DHD);
  k_zero<<<dim3(4),   dim3(256), 0, stream>>>(snorm, (long)BZ*HH*GG);
  k_zero<<<dim3(4),   dim3(256), 0, stream>>>(sesum, (long)BZ*DD);

  // 1) spatial bias MLP, 2) LN1 -> bf16
  k_sb<<<dim3(512), dim3(256), 0, stream>>>(raw_xy, Wsb1, bsb1, Wsb2, bsb2, Wsb3, bsb3, sb);
  k_ln_in<<<dim3(16384), dim3(256), 0, stream>>>(fx, ln1_g, ln1_b, hbf);

  // 3) fused slice: projections + logits + softmax + tok einsum
  k_slice<<<dim3(512), dim3(256), 0, stream>>>(hbf, wx_p, b_x, wfx_p, b_fx,
                                               wsl_p, b_slice, temperat, sb,
                                               swbuf, tok, snorm);
  // 4) token attention, 5) scatter back
  k_attn<<<dim3(BZ), dim3(256), 0, stream>>>(tok, snorm, Wq, Wk, Wv, attn_sc, slice_res, ot);
  k_scatter<<<dim3(4096), dim3(256), 0, stream>>>(swbuf, ot, ybuf);

  // 6) out-proj + residual + LN1p + LN2
  k_outproj<<<dim3(4096), dim3(256), 0, stream>>>(ybuf, wo_p, b_out, fx,
                                                  ln1p_g, ln1p_b, ln2_g, ln2_b,
                                                  fx2, minbuf);
  // 7) fused MLP + residual + LN2p (+SE sums)
  k_mlp<<<dim3(4096), dim3(256), 0, stream>>>(minbuf, wm1_p, bm1, wm2_p, bm2,
                                              fx2, ln2p_g, ln2p_b, out, sesum);
  // 8) SE gate, 9) scale output in place
  k_se<<<dim3(1), dim3(256), 0, stream>>>(sesum, Wse1, bse1, Wse2, bse2, sebuf);
  k_scale<<<dim3(8192), dim3(256), 0, stream>>>(out, sebuf);
}
